// FocalAwareTemporalAttention_35218731827250
// MI455X (gfx1250) — matrix operand order repaired and checked by reading, not compiled, not run on hardware
//
#include <hip/hip_runtime.h>

typedef __attribute__((ext_vector_type(16))) __bf16 v16bf;
typedef __attribute__((ext_vector_type(8)))  float  v8f;

#define B_  4
#define N_  8
#define D_  256
#define NH_ 8
#define HD_ 32
#define P_  4096   // 64*64

__device__ __forceinline__ unsigned short f2bf(float f) {
  unsigned int u = __float_as_uint(f);
  u += 0x7FFFu + ((u >> 16) & 1u);        // round-to-nearest-even
  return (unsigned short)(u >> 16);
}
__device__ __forceinline__ unsigned int pack2(float a, float b) {
  return (unsigned int)f2bf(a) | ((unsigned int)f2bf(b) << 16);
}
__device__ __forceinline__ float silu_f(float x) { return x / (1.0f + __expf(-x)); }

// ---------------------------------------------------------------------------
// Convert the four D x D f32 weight matrices to packed bf16 pairs in ws.
// Layout per matrix: row-major, 128 u32 per row (256 bf16).
// ---------------------------------------------------------------------------
__global__ void cvt_weights(const float* __restrict__ wq, const float* __restrict__ wk,
                            const float* __restrict__ wv, const float* __restrict__ wo,
                            unsigned int* __restrict__ wbf) {
  int idx = blockIdx.x * blockDim.x + threadIdx.x;   // 0 .. 131071
  const float* s = (idx < 32768) ? wq : (idx < 65536) ? wk : (idx < 98304) ? wv : wo;
  int r = idx & 32767;
  wbf[idx] = pack2(s[2 * r], s[2 * r + 1]);
}

// ---------------------------------------------------------------------------
// Tiny MLPs: focus encoder fe[B,N,D], query weights qw2[B,N] (softmaxed),
// relative-position bias rel[B,NH,N,N].
// ---------------------------------------------------------------------------
__global__ void prep_kernel(const float* __restrict__ fd,
    const float* __restrict__ fe_w1, const float* __restrict__ fe_b1,
    const float* __restrict__ fe_w2, const float* __restrict__ fe_b2,
    const float* __restrict__ rp_w,  const float* __restrict__ rp_b,
    const float* __restrict__ qw_w1, const float* __restrict__ qw_b1,
    const float* __restrict__ qw_w2, const float* __restrict__ qw_b2,
    float* __restrict__ fe_out, float* __restrict__ qw2_out, float* __restrict__ rel_out)
{
  __shared__ float h1[256];
  __shared__ float red[128];
  __shared__ float ql[32];
  const int t = threadIdx.x;

  for (int bn = 0; bn < B_ * N_; ++bn) {
    float f = fd[bn];
    h1[t] = silu_f(f * fe_w1[t] + fe_b1[t]);
    if (t < 128) red[t] = qw_w2[t] * silu_f(f * qw_w1[t] + qw_b1[t]);
    __syncthreads();
    float acc = fe_b2[t];
    for (int d = 0; d < 256; ++d) acc += fe_w2[t * 256 + d] * h1[d];
    fe_out[bn * 256 + t] = acc;
    for (int s2 = 64; s2 > 0; s2 >>= 1) {
      if (t < s2) red[t] += red[t + s2];
      __syncthreads();
    }
    if (t == 0) ql[bn] = red[0] + qw_b2[0];
    __syncthreads();
  }
  if (t < B_) {
    float m = -1e30f;
    for (int j = 0; j < N_; ++j) m = fmaxf(m, ql[t * N_ + j]);
    float e[N_], s = 0.f;
    for (int j = 0; j < N_; ++j) { e[j] = __expf(ql[t * N_ + j] - m); s += e[j]; }
    for (int j = 0; j < N_; ++j) qw2_out[t * N_ + j] = e[j] / s;
  }
  for (int idx = t; idx < B_ * NH_ * N_ * N_; idx += 256) {
    int bb = idx >> 9, r = idx & 511;
    int hh = r >> 6, i = (r >> 3) & 7, j = r & 7;
    float dd = fd[bb * N_ + i] - fd[bb * N_ + j];
    rel_out[idx] = tanhf(rp_w[hh] * dd + rp_b[hh]);
  }
}

// ---------------------------------------------------------------------------
// Fused QKV projection + temporal attention + output projection.
// One workgroup = one (b, tile of 8 spatial positions); 8 frames x 8 positions
// = 64 GEMM columns. LDS (dynamic, 230 KB of the 320 KB/WGP):
//   xsp  : bf16-pair x      [128 dpair][64 col]   32 KB
//   qs/ks/vs : f32          [256 e][64 col]       3*64 KB
//   attp : bf16-pair attended [128 epair][8 pos]   4 KB
//   asum : f32 attn sums    [8 h][8 pos][8 n]      2 KB
// ---------------------------------------------------------------------------
__global__ __launch_bounds__(256) void fused_attn(
    const float* __restrict__ features,
    const unsigned int* __restrict__ wqbf, const unsigned int* __restrict__ wkbf,
    const unsigned int* __restrict__ wvbf, const unsigned int* __restrict__ wobf,
    const float* __restrict__ bq, const float* __restrict__ bk,
    const float* __restrict__ bv, const float* __restrict__ bo,
    const float* __restrict__ fe, const float* __restrict__ qw2,
    const float* __restrict__ rel, const float* __restrict__ tempp,
    float* __restrict__ out, float* __restrict__ spat)
{
  extern __shared__ char smem[];
  unsigned int* xsp  = (unsigned int*)(smem);            // 8192 u32
  float*        qs   = (float*)(smem + 32768);           // 16384 f32
  float*        ks   = (float*)(smem + 98304);
  float*        vs   = (float*)(smem + 163840);
  unsigned int* attp = (unsigned int*)(smem + 229376);   // 1024 u32
  float*        asum = (float*)(smem + 233472);          // 512 f32

  const int t  = threadIdx.x;
  const int b  = blockIdx.x >> 9;          // 4 batches
  const int p0 = (blockIdx.x & 511) * 8;   // 512 position tiles

  // ---- Phase 1: stage x = features + 0.1*fe into LDS as packed bf16 ----
  for (int idx = t; idx < 128 * 64; idx += 256) {
    int dp = idx >> 6, col = idx & 63;
    int n = col >> 3, pp = col & 7;
    int d0 = dp * 2;
    size_t base = ((size_t)(b * N_ + n) * D_ + d0) * P_ + (size_t)(p0 + pp);
    float x0 = features[base]      + 0.1f * fe[(b * N_ + n) * D_ + d0];
    float x1 = features[base + P_] + 0.1f * fe[(b * N_ + n) * D_ + d0 + 1];
    xsp[dp * 64 + col] = pack2(x0, x1);
  }
  __syncthreads();

  const int lane  = t & 31;
  const int wave  = t >> 5;          // 8 waves
  const int lm    = lane & 15;
  const int upper = lane >> 4;       // lane half (K interleave select)
  const int apOff = upper * 8;       // A: u32-pair offset (K halves contiguous)
  const int bpOff = upper * 8;       // B: u32-pair offset (K halves contiguous)

  // ---- Phase 2: Q/K/V GEMMs, 192 WMMA tile jobs across 8 waves ----
  for (int job = wave; job < 192; job += 8) {
    int mat = job % 3;
    int rest = job / 3;
    int etile = rest & 15;
    int ctile = rest >> 4;
    const unsigned int* wrow =
        ((mat == 0) ? wqbf : (mat == 1) ? wkbf : wvbf) + (etile * 16 + lm) * 128;
    const float* bias = (mat == 0) ? bq : (mat == 1) ? bk : bv;
    float* dst = (mat == 0) ? qs : (mat == 1) ? ks : vs;
    int col = ctile * 16 + lm;
    __builtin_prefetch(wrow, 0, 3);
    v8f acc = {};
#pragma unroll
    for (int kb = 0; kb < 256; kb += 32) {
      union { v16bf v; uint4 u4[2]; unsigned int u[8]; } A, Bm;
      const uint4* wp = (const uint4*)(wrow + (kb >> 1) + apOff);
      A.u4[0] = wp[0];              // K = kb+khalf .. +7
      A.u4[1] = wp[1];              // K = kb+khalf+8 .. +15
      int bbase = (kb >> 1) + bpOff;
#pragma unroll
      for (int j = 0; j < 8; ++j) Bm.u[j] = xsp[(bbase + j) * 64 + col];
      acc = __builtin_amdgcn_wmma_f32_16x16x32_bf16(
          false, A.v, false, Bm.v, (short)0, acc, false, false);
    }
#pragma unroll
    for (int r = 0; r < 8; ++r) {
      int e = etile * 16 + r + upper * 8;
      dst[e * 64 + col] = acc[r] + bias[e];
    }
  }
  __syncthreads();

  // ---- Phase 3: per-(position, head) attention over N=8 frames (f32) ----
  if (t < 64) {
    int hh = t >> 3, pp = t & 7;
    float sc[8][8];
#pragma unroll
    for (int i = 0; i < 8; ++i)
#pragma unroll
      for (int j = 0; j < 8; ++j) sc[i][j] = 0.f;
    for (int c = 0; c < HD_; ++c) {
      int row = (hh * HD_ + c) * 64;
      float qv[8], kv[8];
#pragma unroll
      for (int i = 0; i < 8; ++i) { qv[i] = qs[row + i * 8 + pp]; kv[i] = ks[row + i * 8 + pp]; }
#pragma unroll
      for (int i = 0; i < 8; ++i)
#pragma unroll
        for (int j = 0; j < 8; ++j) sc[i][j] += qv[i] * kv[j];
    }
    float invs = 1.0f / (tempp[0] * sqrtf((float)HD_));
    const float* relb = rel + ((b * NH_ + hh) * N_) * N_;
    float cj[8];
#pragma unroll
    for (int j = 0; j < 8; ++j) cj[j] = 0.f;
#pragma unroll
    for (int i = 0; i < 8; ++i) {
      float s[8], m = -1e30f;
#pragma unroll
      for (int j = 0; j < 8; ++j) { s[j] = sc[i][j] * invs + relb[i * 8 + j]; m = fmaxf(m, s[j]); }
      float sum = 0.f;
#pragma unroll
      for (int j = 0; j < 8; ++j) { s[j] = __expf(s[j] - m); sum += s[j]; }
      float rs = 1.0f / sum;
      float qwi = qw2[b * N_ + i];
#pragma unroll
      for (int j = 0; j < 8; ++j) { float a = s[j] * rs; sc[i][j] = a; cj[j] += qwi * a; }
    }
#pragma unroll
    for (int j = 0; j < 8; ++j) {         // sum attn over query frames -> stats
      float su = 0.f;
#pragma unroll
      for (int i = 0; i < 8; ++i) su += sc[i][j];
      asum[(hh * 8 + pp) * 8 + j] = su;
    }
    for (int c = 0; c < HD_; c += 2) {    // attended (frame-collapsed), bf16
      int row0 = (hh * HD_ + c) * 64, row1 = row0 + 64;
      float a0 = 0.f, a1 = 0.f;
#pragma unroll
      for (int j = 0; j < 8; ++j) { a0 += cj[j] * vs[row0 + j * 8 + pp]; a1 += cj[j] * vs[row1 + j * 8 + pp]; }
      attp[((hh * HD_ + c) >> 1) * 8 + pp] = pack2(a0, a1);
    }
  }
  __syncthreads();

  // ---- Phase 5: spatial statistics (mean over heads & query frames) ----
  if (t < 64) {
    int pp = t >> 3, n = t & 7;
    float su = 0.f;
#pragma unroll
    for (int hh = 0; hh < 8; ++hh) su += asum[(hh * 8 + pp) * 8 + n];
    spat[(size_t)(b * N_ + n) * P_ + (size_t)(p0 + pp)] = su * (1.0f / 64.0f);
  }

  // ---- Phase 4: output projection, 16 WMMA tile jobs (8 valid columns) ----
  for (int job = wave; job < 16; job += 8) {
    int etile = job;
    const unsigned int* wrow = wobf + (etile * 16 + lm) * 128;
    v8f acc = {};
#pragma unroll
    for (int kb = 0; kb < 256; kb += 32) {
      union { v16bf v; uint4 u4[2]; unsigned int u[8]; } A, Bm;
      const uint4* wp = (const uint4*)(wrow + (kb >> 1) + apOff);
      A.u4[0] = wp[0];
      A.u4[1] = wp[1];
      int bbase = (kb >> 1) + bpOff;
#pragma unroll
      for (int j = 0; j < 8; ++j) Bm.u[j] = (lm < 8) ? attp[(bbase + j) * 8 + lm] : 0u;
      acc = __builtin_amdgcn_wmma_f32_16x16x32_bf16(
          false, A.v, false, Bm.v, (short)0, acc, false, false);
    }
    if (lm < 8) {
#pragma unroll
      for (int r = 0; r < 8; ++r) {
        int e = etile * 16 + r + upper * 8;
        out[(size_t)(b * D_ + e) * P_ + (size_t)(p0 + lm)] = acc[r] + bo[e];
      }
    }
  }
}

// ---------------------------------------------------------------------------
// frame_weights[b,n] = mean over p of spatial[b,n,p] (fixed-order reduction).
// ---------------------------------------------------------------------------
__global__ void fw_kernel(const float* __restrict__ spat, float* __restrict__ fwout) {
  __shared__ float red[256];
  int bn = blockIdx.x, t = threadIdx.x;
  float s = 0.f;
  for (int p = t; p < P_; p += 256) s += spat[(size_t)bn * P_ + p];
  red[t] = s;
  __syncthreads();
  for (int k = 128; k > 0; k >>= 1) {
    if (t < k) red[t] += red[t + k];
    __syncthreads();
  }
  if (t == 0) fwout[bn] = red[0] * (1.0f / (float)P_);
}

extern "C" void kernel_launch(void* const* d_in, const int* in_sizes, int n_in,
                              void* d_out, int out_size, void* d_ws, size_t ws_size,
                              hipStream_t stream) {
  (void)in_sizes; (void)n_in; (void)out_size; (void)ws_size;
  const float* features = (const float*)d_in[0];
  const float* fd    = (const float*)d_in[1];
  const float* wq    = (const float*)d_in[2];
  const float* bq    = (const float*)d_in[3];
  const float* wk    = (const float*)d_in[4];
  const float* bk    = (const float*)d_in[5];
  const float* wv    = (const float*)d_in[6];
  const float* bv    = (const float*)d_in[7];
  const float* wo    = (const float*)d_in[8];
  const float* bo    = (const float*)d_in[9];
  const float* fe_w1 = (const float*)d_in[10];
  const float* fe_b1 = (const float*)d_in[11];
  const float* fe_w2 = (const float*)d_in[12];
  const float* fe_b2 = (const float*)d_in[13];
  const float* rp_w  = (const float*)d_in[14];
  const float* rp_b  = (const float*)d_in[15];
  const float* qw_w1 = (const float*)d_in[16];
  const float* qw_b1 = (const float*)d_in[17];
  const float* qw_w2 = (const float*)d_in[18];
  const float* qw_b2 = (const float*)d_in[19];
  const float* temp  = (const float*)d_in[20];

  float* out   = (float*)d_out;                       // [B, D, H, W]
  float* fwout = out + (size_t)B_ * D_ * P_;          // [B, N]
  float* spat  = fwout + B_ * N_;                     // [B, N, H, W]

  char* ws = (char*)d_ws;
  unsigned int* wbf  = (unsigned int*)ws;             // 4 * 32768 u32 (bf16 pairs)
  unsigned int* wqbf = wbf;
  unsigned int* wkbf = wbf + 32768;
  unsigned int* wvbf = wbf + 65536;
  unsigned int* wobf = wbf + 98304;
  float* fe  = (float*)(ws + 524288);                 // 8192 f32
  float* qw2 = fe + 8192;                             // 32 f32
  float* rel = qw2 + 32;                              // 2048 f32

  cvt_weights<<<512, 256, 0, stream>>>(wq, wk, wv, wo, wbf);
  prep_kernel<<<1, 256, 0, stream>>>(fd, fe_w1, fe_b1, fe_w2, fe_b2, rp_w, rp_b,
                                     qw_w1, qw_b1, qw_w2, qw_b2, fe, qw2, rel);
  const size_t smemBytes = 235520;                    // 230 KB of 320 KB/WGP
  fused_attn<<<B_ * (P_ / 8), 256, smemBytes, stream>>>(
      features, wqbf, wkbf, wvbf, wobf, bq, bk, bv, bo, fe, qw2, rel, temp, out, spat);
  fw_kernel<<<B_ * N_, 256, 0, stream>>>(spat, fwout);
}
